// E3Gnn_58660663329298
// MI455X (gfx1250) — matrix operand order repaired and checked by reading, not compiled, not run on hardware
//
#include <hip/hip_runtime.h>
#include <hip/hip_bf16.h>
#include <math.h>

// ---------------------------------------------------------------------------
// E3-GNN forward for MI455X (gfx1250): edge MLPs as wave32 WMMA f16 GEMMs.
// Grid = one workgroup per receiver (edges are receiver-major, 511 per node);
// segment sums accumulate in LDS, no global atomics. B-fragment loads are
// software-pipelined (distance-2) and issued as global_load_b128 (explicit
// addrspace(1)) so they tick only LOADcnt, decoupled from LDS DScnt waits.
// Opaque per-iteration weight pointers + a memory barrier keep LICM from
// hoisting B loads across tile iterations (avoids scratch spilling).
// ---------------------------------------------------------------------------

typedef __attribute__((ext_vector_type(16))) _Float16 v16h;
typedef __attribute__((ext_vector_type(8)))  float    v8f;
typedef __attribute__((ext_vector_type(4)))  float    v4f;

#define GASPACE __attribute__((address_space(1)))

#define NNODES 512
#define DEG    511        // edges per receiver
#define NEDGES (NNODES*DEG)
#define FDIM   128
#define KFEAT  288        // 257 padded to 9*32
#define FSTR   296        // feat row stride (f16), 16B-aligned rows
#define ASTR   136        // act  row stride (f16), 16B-aligned rows

union AFrag { v16h v; v4f q[2]; };

__device__ __forceinline__ float siluf(float x) {
  return x * (1.0f / (1.0f + __expf(-x)));
}

// Register-blocked GEMM with distance-2 B prefetch and next-A prefetch.
// D[16 x NT*16] += A[16 x KCH*32] * B. A rows in LDS (stride aStride f16),
// B pre-swizzled fragment-major ((kc*NT+nt)*32+lane)*16 in global memory.
template <int KCH, int NT>
__device__ __forceinline__ void gemm_tiles(const _Float16* aBase, int aStride,
                                           const _Float16* Bsw, v8f* acc,
                                           int lane) {
  const _Float16* rowPtr = aBase + (lane & 15) * aStride;
  const int hi = (lane >> 4) << 3;                 // 0 or 8
  constexpr int TOT = KCH * NT;
  // Explicit global address space: lowers to global_load_b128 (LOADcnt only),
  // even though the pointer was made opaque by inline asm.
  const GASPACE _Float16* B = (const GASPACE _Float16*)Bsw;
  AFrag a[2];
  a[0].q[0] = *(const v4f*)(rowPtr + hi);
  a[0].q[1] = *(const v4f*)(rowPtr + hi + 16);
  v16h b0 = *(const GASPACE v16h*)(B + (size_t)lane * 16);
  v16h b1 = (TOT > 1) ? *(const GASPACE v16h*)(B + ((size_t)32 + lane) * 16) : b0;
#pragma unroll
  for (int idx = 0; idx < TOT; ++idx) {
    const int kc = idx / NT;
    const int nt = idx % NT;
    if (nt == 0 && kc + 1 < KCH) {               // prefetch next A chunk
      a[(kc + 1) & 1].q[0] = *(const v4f*)(rowPtr + (kc + 1) * 32 + hi);
      a[(kc + 1) & 1].q[1] = *(const v4f*)(rowPtr + (kc + 1) * 32 + hi + 16);
    }
    v16h bn = b0;
    if (idx + 2 < TOT)                           // distance-2 B prefetch
      bn = *(const GASPACE v16h*)(B + ((size_t)(idx + 2) * 32 + lane) * 16);
    acc[nt] = __builtin_amdgcn_wmma_f32_16x16x32_f16(
        false, a[kc & 1].v, false, b0, (short)0, acc[nt], false, false);
    b0 = b1;
    b1 = bn;
  }
}

// Store D tiles (+bias, optional silu) back to LDS rows as f16.
__device__ __forceinline__ void store_act(const v8f* acc, int NT,
                                          const float* bias, _Float16* dBase,
                                          int dStride, int lane, int doSilu) {
  const int col = lane & 15;
  const int r0  = (lane >> 4) * 8;
  for (int nt = 0; nt < NT; ++nt) {
    float bv = bias[nt * 16 + col];
    for (int j = 0; j < 8; ++j) {
      float v = acc[nt][j] + bv;
      if (doSilu) v = siluf(v);
      dBase[(r0 + j) * dStride + nt * 16 + col] = (_Float16)v;
    }
  }
}

// ---------------------------------------------------------------------------
// Weight swizzle into WMMA-B fragment order: dst[((kc*NT+nt)*32+lane)*16+j]
// lane<16 : (K = kc*32+(j&7)+16*(j>>3),   N = nt*16+lane)
// lane>=16: (K = kc*32+8+(j&7)+16*(j>>3), N = nt*16+(lane-16))
// we0mode: feats col k -> We0 row (k<256 ? k+1 : k==256 ? 0 : pad)
// ---------------------------------------------------------------------------
__global__ void k_swizzle(const float* __restrict__ W, _Float16* __restrict__ dst,
                          int Ksrc, int KC, int NT, int ldN, int Ncols, int we0mode) {
  int tid = blockIdx.x * blockDim.x + threadIdx.x;
  int total = KC * NT * 512;
  if (tid >= total) return;
  int j = tid & 15;
  int lane = (tid >> 4) & 31;
  int rest = tid >> 9;
  int nt = rest % NT;
  int kc = rest / NT;
  int K = kc * 32 + (j & 7) + ((j >> 3) << 4) + ((lane >= 16) ? 8 : 0);
  int N = nt * 16 + (lane & 15);
  float val = 0.0f;
  if (we0mode) {
    int row = (K < 256) ? (K + 1) : ((K == 256) ? 0 : -1);
    if (row >= 0 && N < Ncols) val = W[row * ldN + N];
  } else {
    if (K < Ksrc && N < Ncols) val = W[K * ldN + N];
  }
  dst[tid] = (_Float16)val;
}

// ---------------------------------------------------------------------------
// Small node-side kernels
// ---------------------------------------------------------------------------
__global__ void k_center(const float* __restrict__ x, float* __restrict__ vec,
                         float* __restrict__ vec0) {
  __shared__ float mean[3];
  if (threadIdx.x < 3) {
    float s = 0.0f;
    for (int n = 0; n < NNODES; ++n) s += x[n * 3 + threadIdx.x];
    mean[threadIdx.x] = s * (1.0f / (float)NNODES);
  }
  __syncthreads();
  int n = threadIdx.x;
  if (n < NNODES) {
    for (int c = 0; c < 3; ++c) {
      float v = x[n * 3 + c] - mean[c];
      vec[n * 3 + c] = v;
      vec0[n * 3 + c] = v;
    }
  }
}

__global__ void k_hf0(const float* __restrict__ h, const float* __restrict__ Wh0,
                      const float* __restrict__ bh0, float* __restrict__ hf) {
  int tid = blockIdx.x * blockDim.x + threadIdx.x;   // 512*128
  int n = tid >> 7, f = tid & 127;
  float acc = bh0[f];
  for (int k = 0; k < 16; ++k) acc += h[n * 16 + k] * Wh0[k * FDIM + f];
  hf[tid] = acc;
}

__global__ void k_prep(const float* __restrict__ hf, _Float16* __restrict__ hf16) {
  int tid = blockIdx.x * blockDim.x + threadIdx.x;   // 65536
  hf16[tid] = (_Float16)hf[tid];
}

__global__ void k_vecupd(float* __restrict__ vec, const float* __restrict__ vacc) {
  int tid = blockIdx.x * blockDim.x + threadIdx.x;   // 1536
  if (tid < NNODES * 3) vec[tid] += vacc[tid] * (1.0f / 511.0f);
}

__global__ void k_ph0(const float* __restrict__ m_i, const float* __restrict__ hf,
                      const float* __restrict__ Wp0, const float* __restrict__ bp0,
                      float* __restrict__ t0) {
  int tid = blockIdx.x * blockDim.x + threadIdx.x;
  int n = tid >> 7, j = tid & 127;
  float acc = bp0[j];
  for (int k = 0; k < FDIM; ++k) acc += m_i[n * FDIM + k] * Wp0[k * FDIM + j];
  for (int k = 0; k < FDIM; ++k) acc += hf[n * FDIM + k] * Wp0[(FDIM + k) * FDIM + j];
  t0[tid] = siluf(acc);
}

__global__ void k_ph1(const float* __restrict__ t0, const float* __restrict__ Wp1,
                      const float* __restrict__ bp1, float* __restrict__ t1) {
  int tid = blockIdx.x * blockDim.x + threadIdx.x;
  int n = tid >> 7, j = tid & 127;
  float acc = bp1[j];
  for (int k = 0; k < FDIM; ++k) acc += t0[n * FDIM + k] * Wp1[k * FDIM + j];
  t1[tid] = acc;
}

__global__ void k_ph2(const float* __restrict__ t1, const float* __restrict__ Wpl,
                      const float* __restrict__ bpl, float* __restrict__ hf) {
  int tid = blockIdx.x * blockDim.x + threadIdx.x;
  int n = tid >> 7, j = tid & 127;
  float acc = bpl[j];
  for (int k = 0; k < FDIM; ++k) acc += t1[n * FDIM + k] * Wpl[k * FDIM + j];
  hf[tid] += acc;
}

__global__ void k_readv(const float* __restrict__ vec, const float* __restrict__ vec0,
                        const float* __restrict__ Wv, float* __restrict__ out) {
  int tid = blockIdx.x * blockDim.x + threadIdx.x;   // 512*6
  if (tid >= NNODES * 6) return;
  int n = tid / 6, o = (tid % 6) / 3, c = tid % 3;
  out[tid] = (vec[n * 3 + c] - vec0[n * 3 + c]) * Wv[o];
}

__global__ void k_readh(const float* __restrict__ hf, const float* __restrict__ Wr,
                        const float* __restrict__ br, float* __restrict__ out) {
  __shared__ float row[FDIM];
  __shared__ float red[32];
  int n = blockIdx.x, t = threadIdx.x;               // 32 threads
  for (int k = t; k < FDIM; k += 32) row[k] = hf[n * FDIM + k];
  __syncthreads();
  float mx = -1e30f;
  for (int k = t; k < FDIM; k += 32) mx = fmaxf(mx, row[k]);
  red[t] = mx;
  __syncthreads();
  if (t == 0) { float m = red[0]; for (int i = 1; i < 32; ++i) m = fmaxf(m, red[i]); red[0] = m; }
  __syncthreads();
  float m = red[0];
  float sm = 0.0f;
  for (int k = t; k < FDIM; k += 32) sm += __expf(row[k] - m);
  __syncthreads();
  red[t] = sm;
  __syncthreads();
  if (t == 0) { float s = 0.0f; for (int i = 0; i < 32; ++i) s += red[i]; red[0] = s; }
  __syncthreads();
  float inv = 1.0f / red[0];
  float acc = br[t];
  for (int k = 0; k < FDIM; ++k) acc += __expf(row[k] - m) * inv * Wr[k * 32 + t];
  out[n * 32 + t] = acc;
}

// ---------------------------------------------------------------------------
// Edge kernel: one workgroup per receiver r. 128 threads = 4 waves; each wave
// runs 8 iterations of a 16-edge tile (32 tiles = 511 edges + 1 pad row).
// Segment sums accumulate in LDS (ds_add_f32), one global store at the end.
// ---------------------------------------------------------------------------
__global__ __launch_bounds__(128) void k_edge(
    const float* __restrict__ vec, const _Float16* __restrict__ hf16,
    const int* __restrict__ senders,
    const _Float16* __restrict__ We0g, const _Float16* __restrict__ We1g,
    const _Float16* __restrict__ We2g, const _Float16* __restrict__ Wx0g,
    const _Float16* __restrict__ Wx1g, const _Float16* __restrict__ Winfg,
    const _Float16* __restrict__ Wxog,
    const float* __restrict__ be0, const float* __restrict__ be1,
    const float* __restrict__ be2, const float* __restrict__ bx0,
    const float* __restrict__ bx1, const float* __restrict__ binf,
    const float* __restrict__ bxo,
    float* __restrict__ vacc, float* __restrict__ m_i) {
  __shared__ _Float16 sFeat[4][16][FSTR];   // feats (K=288); later cols 0..127 = m
  __shared__ _Float16 sAct[4][16][ASTR];    // activation ping buffer
  __shared__ _Float16 sHr[FDIM];            // receiver hf row (shared by block)
  __shared__ float sDv[4][16][3];
  __shared__ float sLen[4][16];
  __shared__ float sE[4][16];
  __shared__ float sPx[4][16];
  __shared__ int   sS[4][16];
  __shared__ float sMi[FDIM];               // block-level m_i accumulator
  __shared__ float sSh[4];                  // block-level shift accumulator

  const int w = threadIdx.x >> 5;
  const int lane = threadIdx.x & 31;
  const int r = blockIdx.x;                 // receiver node

  // ---- block init: zero accumulators, cache receiver row ----
  sMi[threadIdx.x] = 0.0f;
  if (threadIdx.x < 4) sSh[threadIdx.x] = 0.0f;
  if (threadIdx.x < 16)
    *(v4f*)&sHr[threadIdx.x * 8] =
        *(const v4f*)(hf16 + r * FDIM + threadIdx.x * 8);
  float rx = vec[r * 3 + 0], ry = vec[r * 3 + 1], rz = vec[r * 3 + 2];
  __syncthreads();

  const _Float16* aFeat = &sFeat[w][0][0];
  const _Float16* aAct  = &sAct[w][0][0];
  const v8f vzero = {0, 0, 0, 0, 0, 0, 0, 0};

  for (int it = 0; it < 8; ++it) {
    // Opaque compiler barrier + per-iteration opaque weight pointers: stop
    // LICM from hoisting/caching the hundreds of B-fragment loads across
    // iterations (would spill to scratch). Loads stay L0/L2-resident anyway.
    asm volatile("" ::: "memory");
    const _Float16* We0s = We0g;  const _Float16* We1s = We1g;
    const _Float16* We2s = We2g;  const _Float16* Wx0s = Wx0g;
    const _Float16* Wx1s = Wx1g;  const _Float16* Winfs = Winfg;
    const _Float16* Wxos = Wxog;
    asm volatile("" : "+s"(We0s), "+s"(We1s), "+s"(We2s), "+s"(Wx0s),
                      "+s"(Wx1s), "+s"(Winfs), "+s"(Wxos));

    const int t0 = (w + 4 * it) * 16;       // first edge slot of this tile

    // ---- staging: metadata, dv, lengths, l2 feature ----
    if (lane < 16) {
      int t = t0 + lane;
      int valid = (t < DEG);
      int s = valid ? senders[r * DEG + t] : r;
      sS[w][lane] = s;
      float dx = rx - vec[s * 3 + 0];
      float dy = ry - vec[s * 3 + 1];
      float dz = rz - vec[s * 3 + 2];
      float l2 = dx * dx + dy * dy + dz * dz + 1e-20f;
      sDv[w][lane][0] = dx; sDv[w][lane][1] = dy; sDv[w][lane][2] = dz;
      sLen[w][lane] = sqrtf(l2);
      sFeat[w][lane][256] = (_Float16)l2;   // feats col 256 <-> We0 row 0
    } else if (it == 0) {
      int i = lane - 16;                    // zero pad cols once (persist)
      for (int k = 257; k < KFEAT; ++k) sFeat[w][i][k] = (_Float16)0.0f;
    }
    __syncthreads();

    // ---- gather hf[senders] -> cols 0..127, hf[r] (LDS) -> 128..255 ----
    for (int tt = 0; tt < 16; ++tt) {
      int c = tt * 32 + lane;
      int i = c >> 5, p = c & 31;
      if (p < 16) {
        int node = sS[w][i];
        *(v4f*)&sFeat[w][i][p * 8] = *(const v4f*)(hf16 + node * FDIM + p * 8);
      } else {
        *(v4f*)&sFeat[w][i][p * 8] = *(const v4f*)&sHr[(p & 15) * 8];
      }
    }
    __syncthreads();

    v8f acc[8];

    // ---- phi_e layer 1 : feats(288) -> 128, silu ----
    for (int q = 0; q < 8; ++q) acc[q] = vzero;
    gemm_tiles<9, 8>(aFeat, FSTR, We0s, acc, lane);
    __syncthreads();
    store_act(acc, 8, be0, &sAct[w][0][0], ASTR, lane, 1);
    __syncthreads();

    // ---- phi_e layer 2 : 128 -> 128, silu ----
    for (int q = 0; q < 8; ++q) acc[q] = vzero;
    gemm_tiles<4, 8>(aAct, ASTR, We1s, acc, lane);
    __syncthreads();
    store_act(acc, 8, be1, &sAct[w][0][0], ASTR, lane, 1);
    __syncthreads();

    // ---- phi_e layer 3 : 128 -> 128, no act; park m in sFeat cols 0..127 ----
    for (int q = 0; q < 8; ++q) acc[q] = vzero;
    gemm_tiles<4, 8>(aAct, ASTR, We2s, acc, lane);
    __syncthreads();
    store_act(acc, 8, be2, &sFeat[w][0][0], FSTR, lane, 0);
    __syncthreads();

    // ---- phi_inf gate : m(128) -> 1, sigmoid (pad row masked to 0) ----
    {
      v8f g = vzero;
      gemm_tiles<4, 1>(aFeat, FSTR, Winfs, &g, lane);
      if ((lane & 15) == 0) {
        int r0 = (lane >> 4) * 8;
        float bb = binf[0];
        for (int j = 0; j < 8; ++j) {
          float z = g[j] + bb;
          float e = 1.0f / (1.0f + __expf(-z));
          sE[w][r0 + j] = (t0 + r0 + j < DEG) ? e : 0.0f;
        }
      }
    }

    // ---- phi_x layer 1 : m(128) -> 128, silu ----
    for (int q = 0; q < 8; ++q) acc[q] = vzero;
    gemm_tiles<4, 8>(aFeat, FSTR, Wx0s, acc, lane);
    __syncthreads();
    store_act(acc, 8, bx0, &sAct[w][0][0], ASTR, lane, 1);
    __syncthreads();

    // ---- phi_x layer 2 : 128 -> 128, silu ----
    for (int q = 0; q < 8; ++q) acc[q] = vzero;
    gemm_tiles<4, 8>(aAct, ASTR, Wx1s, acc, lane);
    __syncthreads();
    store_act(acc, 8, bx1, &sAct[w][0][0], ASTR, lane, 1);
    __syncthreads();

    // ---- phi_x out : 128 -> 1 ----
    {
      v8f g = vzero;
      gemm_tiles<4, 1>(aAct, ASTR, Wxos, &g, lane);
      if ((lane & 15) == 0) {
        int r0 = (lane >> 4) * 8;
        float bb = bxo[0];
        for (int j = 0; j < 8; ++j) sPx[w][r0 + j] = g[j] + bb;
      }
    }
    __syncthreads();

    // ---- epilogue: accumulate segment sums in LDS ----
    {
      int col = lane & 15;
      int r0 = (lane >> 4) * 8;
      for (int nt = 0; nt < 8; ++nt) {
        float p = 0.0f;
        for (int j = 0; j < 8; ++j) {
          int row = r0 + j;
          p += (float)sFeat[w][row][nt * 16 + col] * sE[w][row];
        }
        atomicAdd(&sMi[nt * 16 + col], p);
      }
    }
    if (lane < 16) {
      int row = lane;
      float inv = sPx[w][row] / (1.0f + sLen[w][row]);   // NORM_C = 1
      // pad row: dv == 0 -> zero contribution
      for (int c = 0; c < 3; ++c)
        atomicAdd(&sSh[c], sDv[w][row][c] * inv);
    }
    __syncthreads();
  }

  // ---- write-out: block owns receiver r exclusively (no global atomics) ----
  const float rs = 0.04424339314506803f;    // 1/sqrt(511)
  m_i[r * FDIM + threadIdx.x] = sMi[threadIdx.x] * rs;
  if (threadIdx.x < 3) vacc[r * 3 + threadIdx.x] = sSh[threadIdx.x];
}

// ---------------------------------------------------------------------------
extern "C" void kernel_launch(void* const* d_in, const int* in_sizes, int n_in,
                              void* d_out, int out_size, void* d_ws, size_t ws_size,
                              hipStream_t stream) {
  const float* x        = (const float*)d_in[0];
  const float* h        = (const float*)d_in[1];
  const int*   senders  = (const int*)d_in[2];
  const float* Wh0 = (const float*)d_in[4];
  const float* bh0 = (const float*)d_in[5];
  const float* We0 = (const float*)d_in[6];
  const float* be0 = (const float*)d_in[7];
  const float* We1 = (const float*)d_in[8];
  const float* be1 = (const float*)d_in[9];
  const float* We2 = (const float*)d_in[10];
  const float* be2 = (const float*)d_in[11];
  const float* Winf= (const float*)d_in[12];
  const float* binf= (const float*)d_in[13];
  const float* Wx0 = (const float*)d_in[14];
  const float* bx0 = (const float*)d_in[15];
  const float* Wx1 = (const float*)d_in[16];
  const float* bx1 = (const float*)d_in[17];
  const float* Wxo = (const float*)d_in[18];
  const float* bxo = (const float*)d_in[19];
  const float* Wp0 = (const float*)d_in[20];
  const float* bp0 = (const float*)d_in[21];
  const float* Wp1 = (const float*)d_in[22];
  const float* bp1 = (const float*)d_in[23];
  const float* Wpl = (const float*)d_in[24];
  const float* bpl = (const float*)d_in[25];
  const float* Wv  = (const float*)d_in[26];
  const float* Wr  = (const float*)d_in[27];
  const float* br  = (const float*)d_in[28];
  float* out = (float*)d_out;

  // workspace carve-out (256B aligned)
  char* ws = (char*)d_ws;
  size_t off = 0;
  auto carve = [&](size_t bytes) -> char* {
    char* p = ws + off;
    off = (off + bytes + 255) & ~(size_t)255;
    return p;
  };
  float* f_vec  = (float*)carve(NNODES * 3 * sizeof(float));
  float* f_vec0 = (float*)carve(NNODES * 3 * sizeof(float));
  float* f_vacc = (float*)carve(NNODES * 3 * sizeof(float));
  float* f_hf   = (float*)carve(NNODES * FDIM * sizeof(float));
  float* f_mi   = (float*)carve(NNODES * FDIM * sizeof(float));
  float* f_t0   = (float*)carve(NNODES * FDIM * sizeof(float));
  float* f_t1   = (float*)carve(NNODES * FDIM * sizeof(float));
  _Float16* h_hf16 = (_Float16*)carve(NNODES * FDIM * sizeof(_Float16));
  _Float16* h_We0s[2], *h_We1s[2], *h_We2s[2], *h_Wx0s[2], *h_Wx1s[2], *h_Winfs[2], *h_Wxos[2];
  for (int b = 0; b < 2; ++b) {
    h_We0s[b]  = (_Float16*)carve(9 * 8 * 512 * sizeof(_Float16));
    h_We1s[b]  = (_Float16*)carve(4 * 8 * 512 * sizeof(_Float16));
    h_We2s[b]  = (_Float16*)carve(4 * 8 * 512 * sizeof(_Float16));
    h_Wx0s[b]  = (_Float16*)carve(4 * 8 * 512 * sizeof(_Float16));
    h_Wx1s[b]  = (_Float16*)carve(4 * 8 * 512 * sizeof(_Float16));
    h_Winfs[b] = (_Float16*)carve(4 * 1 * 512 * sizeof(_Float16));
    h_Wxos[b]  = (_Float16*)carve(4 * 1 * 512 * sizeof(_Float16));
  }

  // input embedding + centering
  k_center<<<1, 512, 0, stream>>>(x, f_vec, f_vec0);
  k_hf0<<<256, 256, 0, stream>>>(h, Wh0, bh0, f_hf);

  // weight swizzles (both blocks up front)
  for (int b = 0; b < 2; ++b) {
    k_swizzle<<<144, 256, 0, stream>>>(We0 + (size_t)b * 257 * 128, h_We0s[b], 257, 9, 8, 128, 128, 1);
    k_swizzle<<<64, 256, 0, stream>>>(We1 + (size_t)b * 128 * 128, h_We1s[b], 128, 4, 8, 128, 128, 0);
    k_swizzle<<<64, 256, 0, stream>>>(We2 + (size_t)b * 128 * 128, h_We2s[b], 128, 4, 8, 128, 128, 0);
    k_swizzle<<<64, 256, 0, stream>>>(Wx0 + (size_t)b * 128 * 128, h_Wx0s[b], 128, 4, 8, 128, 128, 0);
    k_swizzle<<<64, 256, 0, stream>>>(Wx1 + (size_t)b * 128 * 128, h_Wx1s[b], 128, 4, 8, 128, 128, 0);
    k_swizzle<<<8, 256, 0, stream>>>(Winf + (size_t)b * 128, h_Winfs[b], 128, 4, 1, 1, 1, 0);
    k_swizzle<<<8, 256, 0, stream>>>(Wxo + (size_t)b * 128, h_Wxos[b], 128, 4, 1, 1, 1, 0);
  }

  for (int b = 0; b < 2; ++b) {
    k_prep<<<256, 256, 0, stream>>>(f_hf, h_hf16);
    k_edge<<<NNODES, 128, 0, stream>>>(
        f_vec, h_hf16, senders,
        h_We0s[b], h_We1s[b], h_We2s[b], h_Wx0s[b], h_Wx1s[b], h_Winfs[b], h_Wxos[b],
        be0 + b * 128, be1 + b * 128, be2 + b * 128,
        bx0 + b * 128, bx1 + b * 128, binf + b, bxo + b,
        f_vacc, f_mi);
    k_vecupd<<<6, 256, 0, stream>>>(f_vec, f_vacc);
    k_ph0<<<256, 256, 0, stream>>>(f_mi, f_hf, Wp0 + (size_t)b * 256 * 128, bp0 + b * 128, f_t0);
    k_ph1<<<256, 256, 0, stream>>>(f_t0, Wp1 + (size_t)b * 128 * 128, bp1 + b * 128, f_t1);
    k_ph2<<<256, 256, 0, stream>>>(f_t1, Wpl + (size_t)b * 128 * 128, bpl + b * 128, f_hf);
  }

  // readout: vout [512,1,2,3] then hout [512,1,32]
  k_readv<<<12, 256, 0, stream>>>(f_vec, f_vec0, Wv, out);
  k_readh<<<512, 32, 0, stream>>>(f_hf, Wr, br, out + NNODES * 6);
}
